// HardBinaryConv_60516089200645
// MI455X (gfx1250) — matrix operand order repaired and checked by reading, not compiled, not run on hardware
//
#include <hip/hip_runtime.h>

// ---------------------------------------------------------------------------
// HardBinaryConv for MI455X (gfx1250, wave32, WMMA).
// im2col GEMM: A (acts -> fp16) x sign(W) (fp16 +-1), quantized every K=64.
// Wave register tile: 2x2 (M x N) 16x16 tiles -> 8 WMMA per quant tile.
// Block: 256 thr = 8 waves, M=32 spatial x N=256 channels.
// Double-buffered LDS A panel, one barrier per quant tile.
// x is pre-padded to fp16 [B][C][58][58] in workspace (if it fits) so the
// hot-loop im2col gather is unconditional 16-bit loads with no bounds math.
// ---------------------------------------------------------------------------

typedef _Float16 half8  __attribute__((ext_vector_type(8)));
typedef _Float16 half16 __attribute__((ext_vector_type(16)));
typedef float    f32x8  __attribute__((ext_vector_type(8)));

#define IN_CHN  256
#define OUT_CHN 256
#define KDIM    2304                 // 256 * 3 * 3
#define IMG_H   56
#define IMG_W   56
#define LSPAT   (IMG_H * IMG_W)      // 3136
#define PW      58                   // padded width/height
#define PSLICE  (PW * PW)            // 3364 halfs per (b, ci)
#define BATCH   4
#define NTILES  36                   // 2304 / 64 quant tiles
#define LDS_STRIDE 72                // 64 + 8 halfs pad (breaks bank conflicts)
#define PANEL   (32 * LDS_STRIDE)    // one A panel, in halfs

// ---------------------------------------------------------------------------
// Kernel 1: scale[c] = mean|W_c|, bsign = sign(W) as fp16, row-major [Cout][K]
// ---------------------------------------------------------------------------
__global__ __launch_bounds__(256) void prep_weights_kernel(
    const float* __restrict__ w, float* __restrict__ scale,
    _Float16* __restrict__ bsign) {
  __shared__ float red[256];
  const int c = blockIdx.x;
  const int t = threadIdx.x;
  const float* wr = w + (size_t)c * KDIM;
  _Float16* br = bsign + (size_t)c * KDIM;
  float s = 0.f;
#pragma unroll
  for (int i = 0; i < 9; ++i) {
    const int k = t + i * 256;
    const float v = wr[k];
    s += fabsf(v);
    br[k] = (_Float16)((v > 0.f) ? 1.f : (v < 0.f ? -1.f : 0.f));
  }
  red[t] = s;
  __syncthreads();
  for (int off = 128; off > 0; off >>= 1) {
    if (t < off) red[t] += red[t + off];
    __syncthreads();
  }
  if (t == 0) scale[c] = red[0] * (1.f / (float)KDIM);
}

// ---------------------------------------------------------------------------
// Kernel 1b: zero-padded fp16 copy of x: xpad[b][ci][1+ih][1+iw] = x[...]
// ---------------------------------------------------------------------------
__global__ __launch_bounds__(256) void pad_x_kernel(
    const float* __restrict__ x, _Float16* __restrict__ xpad) {
  const int ci = blockIdx.x;
  const int b  = blockIdx.y;
  const float* xs = x + ((size_t)b * IN_CHN + ci) * LSPAT;
  _Float16* xd = xpad + ((size_t)b * IN_CHN + ci) * PSLICE;
  for (int e = threadIdx.x; e < PSLICE; e += 256) {
    const int ih = e / PW;
    const int iw = e - ih * PW;
    float v = 0.f;
    if (ih >= 1 && ih <= IMG_H && iw >= 1 && iw <= IMG_W)
      v = xs[(ih - 1) * IMG_W + (iw - 1)];
    xd[e] = (_Float16)v;
  }
}

// ---------------------------------------------------------------------------
// Kernel 2: main WMMA GEMM (PADDED: unconditional fp16 gathers).
// ---------------------------------------------------------------------------
template <bool PADDED>
__global__ __launch_bounds__(256) void binconv_wmma_kernel(
    const float* __restrict__ x, const _Float16* __restrict__ xpad,
    const _Float16* __restrict__ bsign, const float* __restrict__ scale,
    float* __restrict__ out) {
  __shared__ _Float16 ldsA[2 * PANEL];           // 9 KB, ping-pong A panels

  const int mbase = blockIdx.x * 32;             // spatial tile base
  const int b     = blockIdx.z;
  const int tid   = threadIdx.x;
  const int lane  = tid & 31;
  const int wave  = tid >> 5;
  const int lo16  = lane & 15;
  const int hi    = (lane >> 4) & 1;             // 0: lanes 0-15, 1: 16-31

  const int ncol0 = wave * 32 + lo16;            // wave owns cols [w*32,+32)
  const int ncol1 = ncol0 + 16;

  const float*    xb  = x    + (size_t)b * IN_CHN * LSPAT;
  const _Float16* xpb = xpad + (size_t)b * IN_CHN * PSLICE;

  // ---- staging geometry (thread-invariant): k = tid&63, rows mth+4i -------
  const int kth = tid & 63;
  const int mth = tid >> 6;                      // 0..3
  int ohv[8], owv[8], pos[8];
#pragma unroll
  for (int i = 0; i < 8; ++i) {
    const int l  = mbase + mth + 4 * i;
    const int oh = l / IMG_W;
    const int ow = l - oh * IMG_W;
    if constexpr (PADDED) {
      pos[i] = oh * PW + ow;                     // (oh+ky)*58 + (ow+kx) - base
    } else {
      ohv[i] = oh; owv[i] = ow;
    }
  }

  // im2col gather of one K=64 panel row-set for tile T
  auto gather = [&](int T, _Float16 vh[8], float vf[8]) {
    const int kg = T * 64 + kth;
    const int ci = kg / 9;
    const int r9 = kg - ci * 9;
    const int ky = r9 / 3;
    const int kx = r9 - ky * 3;
    if constexpr (PADDED) {
      const _Float16* rowb = xpb + (size_t)ci * PSLICE + ky * PW + kx;
#pragma unroll
      for (int i = 0; i < 8; ++i) vh[i] = rowb[pos[i]];   // global_load d16
    } else {
      const float* xc = xb + (size_t)ci * LSPAT + (ky - 1) * IMG_W + (kx - 1);
#pragma unroll
      for (int i = 0; i < 8; ++i) {
        const int ih = ohv[i] + ky - 1;
        const int iw = owv[i] + kx - 1;
        vf[i] = 0.f;
        if ((unsigned)ih < (unsigned)IMG_H && (unsigned)iw < (unsigned)IMG_W)
          vf[i] = xc[ohv[i] * IMG_W + owv[i]];
      }
    }
  };
  auto commit = [&](_Float16* buf, const _Float16 vh[8], const float vf[8]) {
#pragma unroll
    for (int i = 0; i < 8; ++i)
      buf[(mth + 4 * i) * LDS_STRIDE + kth] =
          PADDED ? vh[i] : (_Float16)vf[i];      // ds_store_b16
  };

  // ---- fragment source pointers (ISA 7.12.2 layouts) ----------------------
  // A 16x32 f16: lane<16 -> M=lane, K 0-7 & 16-23; lane>=16 -> M=lane-16, K+8
  const int aoff0 = (0 * 16 + lo16) * LDS_STRIDE + hi * 8;
  const int aoff1 = (1 * 16 + lo16) * LDS_STRIDE + hi * 8;
  // B 32x16 f16: lane<16 -> K 0-7 & 16-23; lane>=16 -> K+8, N = lane&15
  const _Float16* bB0 = bsign + (size_t)ncol0 * KDIM + hi * 8;
  const _Float16* bB1 = bsign + (size_t)ncol1 * KDIM + hi * 8;

  f32x8 cacc[2][2] = {};                         // per-quant-tile fp32 psums
  f32x8 acc[2][2]  = {};                         // quantized running sums

  // ---- prologue: stage tile 0 into panel 0 --------------------------------
  {
    _Float16 vh[8]; float vf[8];
    gather(0, vh, vf);
    commit(&ldsA[0], vh, vf);
  }
  __syncthreads();

#pragma unroll 1
  for (int t = 0; t < NTILES; ++t) {
    const int p = t & 1;
    _Float16* cur = &ldsA[p * PANEL];
    _Float16* nxt = &ldsA[(1 - p) * PANEL];

    // ---- issue next tile's gathers first: latency hides under WMMAs ------
    _Float16 vh[8]; float vf[8];
    const bool have_next = (t + 1 < NTILES);
    if (have_next) gather(t + 1, vh, vf);

    if (have_next) {                             // global_prefetch_b8 next B
      __builtin_prefetch(bB0 + (size_t)(t + 1) * 64, 0, 3);
      __builtin_prefetch(bB1 + (size_t)(t + 1) * 64, 0, 3);
    }

    // ---- 2 k-steps x 2x2 register tile = 8 WMMAs this window --------------
#pragma unroll
    for (int kk = 0; kk < 2; ++kk) {
      union { half16 v; half8 h[2]; } a0, a1, b0, b1;
      const _Float16* ap0 = cur + aoff0 + kk * 32;
      const _Float16* ap1 = cur + aoff1 + kk * 32;
      a0.h[0] = *(const half8*)(ap0);
      a0.h[1] = *(const half8*)(ap0 + 16);
      a1.h[0] = *(const half8*)(ap1);
      a1.h[1] = *(const half8*)(ap1 + 16);
      const size_t kglob = (size_t)t * 64 + kk * 32;
      const _Float16* bp0 = bB0 + kglob;
      const _Float16* bp1 = bB1 + kglob;
      b0.h[0] = *(const half8*)(bp0);
      b0.h[1] = *(const half8*)(bp0 + 16);
      b1.h[0] = *(const half8*)(bp1);
      b1.h[1] = *(const half8*)(bp1 + 16);

      cacc[0][0] = __builtin_amdgcn_wmma_f32_16x16x32_f16(
          false, a0.v, false, b0.v, (short)0, cacc[0][0], false, false);
      cacc[0][1] = __builtin_amdgcn_wmma_f32_16x16x32_f16(
          false, a0.v, false, b1.v, (short)0, cacc[0][1], false, false);
      cacc[1][0] = __builtin_amdgcn_wmma_f32_16x16x32_f16(
          false, a1.v, false, b0.v, (short)0, cacc[1][0], false, false);
      cacc[1][1] = __builtin_amdgcn_wmma_f32_16x16x32_f16(
          false, a1.v, false, b1.v, (short)0, cacc[1][1], false, false);
    }

    // ---- quantize K=64 psum: rint(med3(psum/16,-16,15)), accumulate ------
#pragma unroll
    for (int mt = 0; mt < 2; ++mt)
#pragma unroll
      for (int nt = 0; nt < 2; ++nt) {
#pragma unroll
        for (int i = 0; i < 8; ++i) {
          const float q =
              __builtin_amdgcn_fmed3f(cacc[mt][nt][i] * 0.0625f, -16.f, 15.f);
          acc[mt][nt][i] += __builtin_rintf(q);  // round-half-even = jnp.round
        }
        const f32x8 z = {};
        cacc[mt][nt] = z;
      }

    // ---- commit staged tile t+1 into the other panel ----------------------
    if (have_next) commit(nxt, vh, vf);
    __syncthreads();  // staged writes visible; prev readers of 'nxt' are past
  }

  // ---- epilogue: OA int8 wrap, * 2^4, * per-channel scale ------------------
  const float sc0 = scale[ncol0] * 16.f;
  const float sc1 = scale[ncol1] * 16.f;
#pragma unroll
  for (int mt = 0; mt < 2; ++mt)
#pragma unroll
    for (int nt = 0; nt < 2; ++nt) {
      const int   nc = nt ? ncol1 : ncol0;
      const float sc = nt ? sc1 : sc0;
      // C/D layout: VGPR i -> M = i + 8*hi, lane&15 -> N
      float* op = out + ((size_t)b * OUT_CHN + nc) * LSPAT
                      + mbase + mt * 16 + hi * 8;
#pragma unroll
      for (int i = 0; i < 8; ++i) {
        const float a = acc[mt][nt][i];          // exact integer in [-576,540]
        float r;
        if (a >= -128.f && a <= 127.f) {
          r = a;
        } else {
          int wv = ((int)a) & 255;               // trunc-toward-zero cast
          if (wv > 127) wv -= 256;
          r = (float)wv;
        }
        op[i] = r * sc;                          // per-lane contiguous 32B
      }
    }
}

// ---------------------------------------------------------------------------
// Launcher. d_in[0] = x (4*256*56*56 f32), d_in[1] = weights (589824 f32).
// Workspace: [0,1KB) scale | [1KB, +1.125MB) bsign fp16 | [, +6.9MB) xpad fp16
// ---------------------------------------------------------------------------
extern "C" void kernel_launch(void* const* d_in, const int* in_sizes, int n_in,
                              void* d_out, int out_size, void* d_ws, size_t ws_size,
                              hipStream_t stream) {
  const float* x = (const float*)d_in[0];
  const float* w = (const float*)d_in[1];
  float* out = (float*)d_out;

  const size_t off_bsign = 1024;
  const size_t sz_bsign  = (size_t)OUT_CHN * KDIM * sizeof(_Float16);
  const size_t off_xpad  = off_bsign + sz_bsign;          // 64B-aligned
  const size_t sz_xpad   = (size_t)BATCH * IN_CHN * PSLICE * sizeof(_Float16);
  const size_t need      = off_xpad + sz_xpad;            // ~7.7 MB

  float* scale = (float*)d_ws;
  _Float16* bsign = (_Float16*)((char*)d_ws + off_bsign);
  _Float16* xpad  = (_Float16*)((char*)d_ws + off_xpad);

  prep_weights_kernel<<<OUT_CHN, 256, 0, stream>>>(w, scale, bsign);

  dim3 grid(LSPAT / 32, 1, BATCH);               // (98, 1, 4)
  if (ws_size >= need) {
    pad_x_kernel<<<dim3(IN_CHN, BATCH), 256, 0, stream>>>(x, xpad);
    binconv_wmma_kernel<true><<<grid, 256, 0, stream>>>(x, xpad, bsign, scale, out);
  } else {
    binconv_wmma_kernel<false><<<grid, 256, 0, stream>>>(x, xpad, bsign, scale, out);
  }
}
